// Attention_module_17179869882
// MI455X (gfx1250) — compile-verified, matrix operand order checked by reading, not run on hardware
//
#include <hip/hip_runtime.h>

#define BB 64
#define SS 2048
#define EE 512
#define DDIM 512
#define NEGV -1e10f

typedef __attribute__((ext_vector_type(16))) __bf16 v16bf;
typedef __attribute__((ext_vector_type(8)))  float  v8f;
typedef __attribute__((ext_vector_type(4)))  unsigned int u32x4;

struct Frag { u32x4 lo, hi; };

static __device__ __forceinline__ unsigned short f2bf(float f) {
  unsigned u = __builtin_bit_cast(unsigned, f);
  u += 0x7FFFu + ((u >> 16) & 1u);           // round-to-nearest-even
  return (unsigned short)(u >> 16);
}

// Async global->LDS 16-byte copy (CDNA5, ASYNCcnt-tracked, no VGPR data path).
static __device__ __forceinline__ void async_copy_b128(unsigned lds_off, const void* gaddr) {
  asm volatile("global_load_async_to_lds_b128 %0, %1, off"
               :: "v"(lds_off), "v"(gaddr) : "memory");
}
static __device__ __forceinline__ void async_wait0() {
  asm volatile("s_wait_asynccnt 0x0" ::: "memory");
}

// ---- K0: Wet[n][k] = bf16(We[k][n])  (one-time convert + transpose, 512KB) ----
__global__ __launch_bounds__(256) void wconv_kernel(
    const float* __restrict__ attn_w, unsigned short* __restrict__ wet)
{
  const int idx = blockIdx.x * 256 + threadIdx.x;   // 0 .. 512*512-1
  const int n = idx >> 9, k = idx & 511;
  wet[idx] = f2bf(attn_w[(long)(DDIM + k) * DDIM + n]);
}

// ---------------- K1: hwhb[b,d] = h[b,:] @ Wh[:,d] + attn_b[d] ----------------
__global__ __launch_bounds__(256) void hwh_kernel(
    const float* __restrict__ hidden, const float* __restrict__ attn_w,
    const float* __restrict__ attn_b, float* __restrict__ hwhb)
{
  const int d = blockIdx.x * 256 + threadIdx.x;
  const int b = blockIdx.y;
  const float* h = hidden + (long)BB * DDIM + (long)b * DDIM;  // hidden[-1]
  float acc = attn_b[d];
  for (int k = 0; k < DDIM; ++k)
    acc = fmaf(h[k], attn_w[(long)k * DDIM + d], acc);
  hwhb[(long)b * DDIM + d] = acc;
}

// ---- K2: scores[b,s] = sum_d tanh( enc[b,s,:]@We[:,d] + hwhb[b,d] ) * v[d] ----
// 8 wave32 per workgroup; 32(M) x 512(N) fp32 tile via v_wmma_f32_16x16x32_bf16.
// Each wave: 2 M-tiles x 4 N-tiles. B staged by double-buffered async-LDS DMA
// overlapped with the matrix pipe; one barrier per k-step.
__global__ __launch_bounds__(256) void energy_scores_kernel(
    const float* __restrict__ enc, const unsigned short* __restrict__ wet,
    const float* __restrict__ hwhb, const float* __restrict__ v,
    float* __restrict__ scores)
{
  __shared__ unsigned short As[2][32][40];    // enc tile (m x k), 80B rows
  __shared__ unsigned short Bs[2][512][40];   // Wet slice (n x k), 80B rows
  __shared__ float HB[DDIM];
  __shared__ float VV[DDIM];
  __shared__ float SC[32];

  const int t  = threadIdx.x;
  const int b  = blockIdx.y;
  const int s0 = blockIdx.x * 32;

  for (int i = t; i < DDIM; i += 256) { HB[i] = hwhb[(long)b * DDIM + i]; VV[i] = v[i]; }
  if (t < 32) SC[t] = 0.f;

  const int wave   = t >> 5, lane = t & 31;
  const int n_base = wave * 4;               // 4 consecutive 16-col N tiles per wave
  const int lrow   = lane & 15;
  const int bk     = (lane < 16) ? 0 : 8;    // ISA 16-bit A/B frag: half-lane K split

  v8f zero = {};
  v8f acc[2][4];
#pragma unroll
  for (int mt = 0; mt < 2; ++mt)
#pragma unroll
    for (int j = 0; j < 4; ++j) acc[mt][j] = zero;

  const int am  = t >> 3;                    // 0..31
  const int akq = (t & 7) * 4;               // 0..28

  // Per-thread share of the B-tile async DMA (8 x b128 = 32KB / 256 threads).
  auto issueB = [&](int k0, int buf) {
#pragma unroll
    for (int i = 0; i < 8; ++i) {
      const int c  = t + 256 * i;            // 0..2047 16-byte chunks
      const int n  = c >> 2;                 // 0..511
      const int kc = (c & 3) * 8;            // halfword offset 0,8,16,24
      async_copy_b128((unsigned)(size_t)&Bs[buf][n][kc],
                      wet + (long)n * EE + k0 + kc);
    }
  };

  issueB(0, 0);                              // prologue: prefetch first B slice

  for (int k0 = 0, it = 0; k0 < EE; k0 += 32, ++it) {
    const int buf = it & 1;
    // A tile: 32x32 fp32 -> bf16 LDS (coalesced float4 loads + convert)
    {
      const float4 f = *reinterpret_cast<const float4*>(
          enc + ((long)b * SS + s0 + am) * EE + k0 + akq);
      unsigned p0 = (unsigned)f2bf(f.x) | ((unsigned)f2bf(f.y) << 16);
      unsigned p1 = (unsigned)f2bf(f.z) | ((unsigned)f2bf(f.w) << 16);
      *reinterpret_cast<unsigned*>(&As[buf][am][akq])     = p0;
      *reinterpret_cast<unsigned*>(&As[buf][am][akq + 2]) = p1;
    }
    async_wait0();                 // B(buf) issued last iteration has landed
    __syncthreads();               // A(buf) stores + everyone done reading buf^1
    if (k0 + 32 < EE) issueB(k0 + 32, buf ^ 1);   // overlap DMA with WMMAs

    Frag fa0, fa1;
    fa0.lo = *reinterpret_cast<const u32x4*>(&As[buf][lrow][bk]);
    fa0.hi = *reinterpret_cast<const u32x4*>(&As[buf][lrow][bk + 16]);
    fa1.lo = *reinterpret_cast<const u32x4*>(&As[buf][16 + lrow][bk]);
    fa1.hi = *reinterpret_cast<const u32x4*>(&As[buf][16 + lrow][bk + 16]);
    v16bf a0 = __builtin_bit_cast(v16bf, fa0);
    v16bf a1 = __builtin_bit_cast(v16bf, fa1);
#pragma unroll
    for (int j = 0; j < 4; ++j) {
      const int nn = (n_base + j) * 16 + lrow;
      Frag fb;
      fb.lo = *reinterpret_cast<const u32x4*>(&Bs[buf][nn][bk]);
      fb.hi = *reinterpret_cast<const u32x4*>(&Bs[buf][nn][bk + 16]);
      v16bf bmat = __builtin_bit_cast(v16bf, fb);
      acc[0][j] = __builtin_amdgcn_wmma_f32_16x16x32_bf16(
          false, a0, false, bmat, (short)0, acc[0][j], false, false);
      acc[1][j] = __builtin_amdgcn_wmma_f32_16x16x32_bf16(
          false, a1, false, bmat, (short)0, acc[1][j], false, false);
    }
  }

  // Epilogue: tanh(acc + hwhb[n]) * v[n], reduce over N.
  // C/D layout: lane%16 = N col, VGPR r = M row (lanes>=16: row r+8).
  float p[2][8];
#pragma unroll
  for (int mt = 0; mt < 2; ++mt)
#pragma unroll
    for (int r = 0; r < 8; ++r) p[mt][r] = 0.f;
#pragma unroll
  for (int j = 0; j < 4; ++j) {
    const int nn = (n_base + j) * 16 + lrow;
    const float hb = HB[nn];
    const float vv = VV[nn];
#pragma unroll
    for (int mt = 0; mt < 2; ++mt)
#pragma unroll
      for (int r = 0; r < 8; ++r)
        p[mt][r] += tanhf(acc[mt][j][r] + hb) * vv;
  }
#pragma unroll
  for (int mt = 0; mt < 2; ++mt)
#pragma unroll
    for (int r = 0; r < 8; ++r) {
      p[mt][r] += __shfl_xor(p[mt][r], 1, 32);
      p[mt][r] += __shfl_xor(p[mt][r], 2, 32);
      p[mt][r] += __shfl_xor(p[mt][r], 4, 32);
      p[mt][r] += __shfl_xor(p[mt][r], 8, 32);
    }
  if (lrow == 0) {
#pragma unroll
    for (int mt = 0; mt < 2; ++mt)
#pragma unroll
      for (int r = 0; r < 8; ++r)
        atomicAdd(&SC[mt * 16 + r + (lane >> 4) * 8], p[mt][r]);
  }
  __syncthreads();
  if (t < 32) scores[(long)b * SS + s0 + t] = SC[t];
}

// ---------------- K3: masked softmax over S per batch row ----------------
__global__ __launch_bounds__(256) void softmax_kernel(
    const float* __restrict__ scores, const int* __restrict__ lens,
    float* __restrict__ attn)
{
  __shared__ float red[256];
  const int b = blockIdx.x, t = threadIdx.x;
  const int len = lens[b];
  float vals[8];
  float m = -3.4e38f;
#pragma unroll
  for (int i = 0; i < 8; ++i) {
    const int s = t + i * 256;
    float val = scores[(long)b * SS + s];
    if (s >= len) val = NEGV;
    vals[i] = val;
    m = fmaxf(m, val);
  }
  red[t] = m; __syncthreads();
  for (int off = 128; off > 0; off >>= 1) {
    if (t < off) red[t] = fmaxf(red[t], red[t + off]);
    __syncthreads();
  }
  m = red[0]; __syncthreads();
  float sum = 0.f;
#pragma unroll
  for (int i = 0; i < 8; ++i) { vals[i] = __expf(vals[i] - m); sum += vals[i]; }
  red[t] = sum; __syncthreads();
  for (int off = 128; off > 0; off >>= 1) {
    if (t < off) red[t] += red[t + off];
    __syncthreads();
  }
  const float inv = 1.f / red[0];
#pragma unroll
  for (int i = 0; i < 8; ++i)
    attn[(long)b * SS + t + i * 256] = vals[i] * inv;
}

// ---------------- K4: context[b,e] = sum_s attn[b,s] * enc[b,s,e] ----------------
__global__ __launch_bounds__(256) void context_kernel(
    const float* __restrict__ enc, const float* __restrict__ attn,
    float* __restrict__ ctx)
{
  const int e = blockIdx.x * 256 + threadIdx.x;
  const int b = blockIdx.y;
  const float* encb = enc + (long)b * SS * EE;
  const float* w    = attn + (long)b * SS;
  float acc = 0.f;
  for (int s = 0; s < SS; ++s)
    acc = fmaf(w[s], encb[(long)s * EE + e], acc);
  ctx[(long)b * EE + e] = acc;
}

extern "C" void kernel_launch(void* const* d_in, const int* in_sizes, int n_in,
                              void* d_out, int out_size, void* d_ws, size_t ws_size,
                              hipStream_t stream)
{
  const float* hidden = (const float*)d_in[0];
  const float* enc    = (const float*)d_in[1];
  const int*   lens   = (const int*)  d_in[2];
  const float* attn_w = (const float*)d_in[3];
  const float* attn_b = (const float*)d_in[4];
  const float* v      = (const float*)d_in[5];

  float* ctx_out  = (float*)d_out;                   // B*E (context, B x 1 x E)
  float* attn_out = (float*)d_out + (long)BB * EE;   // B*S (attention weights)

  float*          hwhb   = (float*)d_ws;                       // B*D fp32
  float*          scores = hwhb + (long)BB * DDIM;             // B*S fp32
  unsigned short* wet    = (unsigned short*)(scores + (long)BB * SS); // E*D bf16

  wconv_kernel<<<dim3((EE * DDIM) / 256), 256, 0, stream>>>(attn_w, wet);
  hwh_kernel<<<dim3(DDIM / 256, BB), 256, 0, stream>>>(hidden, attn_w, attn_b, hwhb);
  energy_scores_kernel<<<dim3(SS / 32, BB), 256, 0, stream>>>(enc, wet, hwhb, v, scores);
  softmax_kernel<<<BB, 256, 0, stream>>>(scores, lens, attn_out);
  context_kernel<<<dim3(EE / 256, BB), 256, 0, stream>>>(enc, attn_out, ctx_out);
}
